// VariableSelectionNetwork_60361470378663
// MI455X (gfx1250) — compile-verified, matrix operand order checked by reading, not run on hardware
//
#include <hip/hip_runtime.h>
#include <hip/hip_bf16.h>
#include <math.h>

// Problem dims (fixed by reference)
#define B_   64
#define T_   128
#define N_   32
#define D_   64
#define H_   128
#define C_   64
#define BT_  (B_ * T_)      // 8192
#define ND_  (N_ * D_)      // 2048
#define KC_  128            // K-chunk for the flatten GEMM stream
#define LN_EPS 1e-5f

typedef _Float16 v16h __attribute__((ext_vector_type(16)));
typedef _Float16 h8   __attribute__((ext_vector_type(8)));
typedef float    v8f  __attribute__((ext_vector_type(8)));

// ---------------------------------------------------------------------------
// CDNA5 async global->LDS staging (GLOBAL_LOAD_ASYNC_TO_LDS_B128, ASYNCcnt).
// GV addressing: vdst = LDS byte address, vaddr = 64-bit global address.
// ---------------------------------------------------------------------------
__device__ __forceinline__ void cp16(void* ldst, const void* gsrc) {
  asm volatile("global_load_async_to_lds_b128 %0, %1, off"
               :
               : "v"((unsigned)(size_t)ldst), "v"((unsigned long long)(size_t)gsrc)
               : "memory");
}
__device__ __forceinline__ void wait_async() {
  asm volatile("s_wait_asynccnt 0x0" ::: "memory");
  __syncthreads();
}

// Load a 16-half WMMA operand fragment. p points at (index, k0 + khalf*8) in a
// row-major [rows][ldK] f16 array; elements 0..7 = K+0..7, 8..15 = K+16..23
// (CDNA5 16-bit A/B VGPR layout, ISA 7.12.2). 16-byte aligned.
__device__ __forceinline__ v16h load_frag16(const _Float16* p) {
  h8 lo = *(const h8*)(p);
  h8 hi = *(const h8*)(p + 16);
  v16h f;
#pragma unroll
  for (int i = 0; i < 8; ++i) { f[i] = lo[i]; f[8 + i] = hi[i]; }
  return f;
}

#define WMMA16(a, b, c) \
  __builtin_amdgcn_wmma_f32_16x16x32_f16(false, (a), false, (b), (short)0, (c), false, false)

__device__ __forceinline__ float elu_f(float x)     { return x > 0.0f ? x : __expf(x) - 1.0f; }
__device__ __forceinline__ float sigmoid_f(float x) { return 1.0f / (1.0f + __expf(-x)); }

// ---------------------------------------------------------------- converts
__global__ void __launch_bounds__(256) k_cvt(const float* __restrict__ src,
                                             _Float16* __restrict__ dst, int n) {
  for (int i = blockIdx.x * blockDim.x + threadIdx.x; i < n; i += gridDim.x * blockDim.x)
    dst[i] = (_Float16)src[i];
}

// src: fp32 [batch][R][C]  ->  dst: f16 [batch][C][R]  (transpose per batch)
__global__ void __launch_bounds__(256) k_cvtT(const float* __restrict__ src,
                                              _Float16* __restrict__ dst,
                                              int batch, int R, int Ccols) {
  int total = batch * R * Ccols;
  for (int i = blockIdx.x * blockDim.x + threadIdx.x; i < total; i += gridDim.x * blockDim.x) {
    int n = i / (R * Ccols);
    int rem = i - n * (R * Ccols);
    int c = rem / R;
    int r = rem - c * R;
    dst[i] = (_Float16)src[((size_t)n * R + r) * Ccols + c];
  }
}

// ctxh[b][h] = sum_c context[b][c] * fWc[c][h]   (tiny, plain FMA)
__global__ void __launch_bounds__(128) k_ctx(const float* __restrict__ context,
                                             const float* __restrict__ fWc,
                                             float* __restrict__ ctxh) {
  int b = blockIdx.x, h = threadIdx.x;
  float s = 0.0f;
  for (int c = 0; c < C_; ++c) s += context[b * C_ + c] * fWc[c * H_ + h];
  ctxh[b * H_ + h] = s;
}

// ---------------------------------------------------------------- flatten GRN
// One block = 128 rows (8 waves, one 16-row M-tile per wave). K=2048 streamed
// in 16 double-buffered async chunks; skip tiles in registers; GLU + LN +
// softmax via 16-lane shfl_xor butterflies. Produces weights [BT, N_].
__global__ void __launch_bounds__(256)
k_flat(const _Float16* __restrict__ e16,
       const _Float16* __restrict__ fW1T, const _Float16* __restrict__ fWskipT,
       const _Float16* __restrict__ fW2T, const _Float16* __restrict__ fWgT,
       const float* __restrict__ fb1, const float* __restrict__ fbskip,
       const float* __restrict__ fb2, const float* __restrict__ fbg,
       const float* __restrict__ fg, const float* __restrict__ fbv,
       const float* __restrict__ ctxh,
       float* __restrict__ wOut) {
  __shared__ _Float16 bufFlat[2][128 * KC_];  // 2x32KB: flat chunks, then act
  __shared__ _Float16 bufW1[2][128 * KC_];    // 2x32KB: W1 chunks, then W2
  __shared__ _Float16 bufWs[2][N_ * KC_];     // 2x8KB : Wskip chunks, then Wg

  const int tid = threadIdx.x;
  const int wave = tid >> 5, lane = tid & 31;
  const int laneR = lane & 15, kh = (lane >> 4) & 1;
  const int hi8 = (lane & 16) ? 8 : 0;
  const int r0 = wave * 16;
  const int rowbase = blockIdx.x * 128;
  const int bIdx = rowbase / T_;              // T_==128: one batch per block
  const v8f Z = {0.f, 0.f, 0.f, 0.f, 0.f, 0.f, 0.f, 0.f};

  auto stageChunk = [&](int kc, int p) {
    int kk = kc * KC_;
    for (int j = 0; j < 8; ++j) {             // flat rows: 2048 x 16B
      int c = tid + j * 256;
      int r = c >> 4, off = (c & 15) * 8;
      cp16(&bufFlat[p][r * KC_ + off], e16 + (size_t)(rowbase + r) * ND_ + kk + off);
    }
    for (int j = 0; j < 8; ++j) {             // W1 panel: 2048 x 16B
      int c = tid + j * 256;
      int col = c >> 4, off = (c & 15) * 8;
      cp16(&bufW1[p][col * KC_ + off], fW1T + (size_t)col * ND_ + kk + off);
    }
    for (int j = 0; j < 2; ++j) {             // Wskip panel: 512 x 16B
      int c = tid + j * 256;
      int col = c >> 4, off = (c & 15) * 8;
      cp16(&bufWs[p][col * KC_ + off], fWskipT + (size_t)col * ND_ + kk + off);
    }
  };

  // ---- GEMM1 stream: accW = flat @ W1 (8 tiles), accS = flat @ Wskip ----
  v8f accW[8], accS[2];
#pragma unroll
  for (int ct = 0; ct < 8; ++ct) accW[ct] = Z;
  accS[0] = Z; accS[1] = Z;

  stageChunk(0, 0);
  for (int kc = 0; kc < ND_ / KC_; ++kc) {
    int p = kc & 1;
    wait_async();                             // chunk kc resident
    if (kc + 1 < ND_ / KC_) stageChunk(kc + 1, p ^ 1);   // overlaps compute
#pragma unroll
    for (int k0 = 0; k0 < KC_; k0 += 32) {
      v16h a = load_frag16(&bufFlat[p][(r0 + laneR) * KC_ + k0 + kh * 8]);
#pragma unroll
      for (int ct = 0; ct < 8; ++ct) {
        v16h b = load_frag16(&bufW1[p][(ct * 16 + laneR) * KC_ + k0 + kh * 8]);
        accW[ct] = WMMA16(a, b, accW[ct]);
      }
#pragma unroll
      for (int cs = 0; cs < 2; ++cs) {
        v16h b = load_frag16(&bufWs[p][(cs * 16 + laneR) * KC_ + k0 + kh * 8]);
        accS[cs] = WMMA16(a, b, accS[cs]);
      }
    }
    __syncthreads();                          // buffer reusable 2 chunks later
  }

  // ---- stage W2 (-> bufW1[0]) and Wg (-> bufWs, 16KB contiguous) ----
  _Float16* sAct = &bufFlat[0][0];
  _Float16* sW2  = &bufW1[0][0];
  _Float16* sWg  = &bufWs[0][0];
  for (int j = 0; j < 8; ++j) { int c = tid + j * 256; cp16(sW2 + c * 8, fW2T + (size_t)c * 8); }
  for (int j = 0; j < 4; ++j) { int c = tid + j * 256; cp16(sWg + c * 8, fWgT + (size_t)c * 8); }

  // ---- h1 = elu(accW + b1 + ctx) -> sAct (own rows); skip in registers ----
  float sk[2][8];
#pragma unroll
  for (int ct = 0; ct < 8; ++ct) {
    int col = ct * 16 + laneR;
    float bc = fb1[col] + ctxh[bIdx * H_ + col];
#pragma unroll
    for (int i = 0; i < 8; ++i)
      sAct[(r0 + i + hi8) * KC_ + col] = (_Float16)elu_f(accW[ct][i] + bc);
  }
#pragma unroll
  for (int cs = 0; cs < 2; ++cs) {
    float bs = fbskip[cs * 16 + laneR];
#pragma unroll
    for (int i = 0; i < 8; ++i) sk[cs][i] = accS[cs][i] + bs;
  }
  wait_async();                               // W2 + Wg resident

  // ---- GEMM2 (K=128): h2 = h1 @ W2 + b2 (own rows, no barriers) ----
  v8f h2[8];
#pragma unroll
  for (int ct = 0; ct < 8; ++ct) h2[ct] = Z;
#pragma unroll
  for (int k0 = 0; k0 < H_; k0 += 32) {
    v16h a = load_frag16(sAct + (r0 + laneR) * KC_ + k0 + kh * 8);
#pragma unroll
    for (int ct = 0; ct < 8; ++ct) {
      v16h b = load_frag16(sW2 + (ct * 16 + laneR) * H_ + k0 + kh * 8);
      h2[ct] = WMMA16(a, b, h2[ct]);
    }
  }
#pragma unroll
  for (int ct = 0; ct < 8; ++ct) {
    int col = ct * 16 + laneR;
    float b2v = fb2[col];
#pragma unroll
    for (int i = 0; i < 8; ++i)
      sAct[(r0 + i + hi8) * KC_ + col] = (_Float16)(h2[ct][i] + b2v);
  }

  // ---- GEMM3 (K=128, 64 cols): g = h2 @ Wg + bg, 4 tiles in registers ----
  v8f g[4];
#pragma unroll
  for (int ct = 0; ct < 4; ++ct) g[ct] = Z;
#pragma unroll
  for (int k0 = 0; k0 < H_; k0 += 32) {
    v16h a = load_frag16(sAct + (r0 + laneR) * KC_ + k0 + kh * 8);
#pragma unroll
    for (int ct = 0; ct < 4; ++ct) {
      v16h b = load_frag16(sWg + (ct * 16 + laneR) * H_ + k0 + kh * 8);
      g[ct] = WMMA16(a, b, g[ct]);
    }
  }

  // ---- GLU (col c with gate c+32) + skip, all in registers ----
  float t[2][8];
#pragma unroll
  for (int cs = 0; cs < 2; ++cs) {
    int col = cs * 16 + laneR;
    float ba = fbg[col], bg2 = fbg[col + 32];
#pragma unroll
    for (int i = 0; i < 8; ++i)
      t[cs][i] = (g[cs][i] + ba) * sigmoid_f(g[cs + 2][i] + bg2) + sk[cs][i];
  }

  // ---- LayerNorm over N=32 + softmax via 16-lane butterflies ----
  float s1[8], s2[8];
#pragma unroll
  for (int i = 0; i < 8; ++i) {
    s1[i] = t[0][i] + t[1][i];
    s2[i] = t[0][i] * t[0][i] + t[1][i] * t[1][i];
  }
#pragma unroll
  for (int m = 1; m < 16; m <<= 1)
#pragma unroll
    for (int i = 0; i < 8; ++i) {
      s1[i] += __shfl_xor(s1[i], m, 32);
      s2[i] += __shfl_xor(s2[i], m, 32);
    }
  float l[2][8];
#pragma unroll
  for (int i = 0; i < 8; ++i) {
    float mu = s1[i] * (1.0f / N_);
    float var = s2[i] * (1.0f / N_) - mu * mu;
    float rs = rsqrtf(var + LN_EPS);
#pragma unroll
    for (int cs = 0; cs < 2; ++cs) {
      int col = cs * 16 + laneR;
      l[cs][i] = (t[cs][i] - mu) * rs * fg[col] + fbv[col];
    }
  }
  float mx[8];
#pragma unroll
  for (int i = 0; i < 8; ++i) mx[i] = fmaxf(l[0][i], l[1][i]);
#pragma unroll
  for (int m = 1; m < 16; m <<= 1)
#pragma unroll
    for (int i = 0; i < 8; ++i) mx[i] = fmaxf(mx[i], __shfl_xor(mx[i], m, 32));
  float e0[8], e1[8], se[8];
#pragma unroll
  for (int i = 0; i < 8; ++i) {
    e0[i] = __expf(l[0][i] - mx[i]);
    e1[i] = __expf(l[1][i] - mx[i]);
    se[i] = e0[i] + e1[i];
  }
#pragma unroll
  for (int m = 1; m < 16; m <<= 1)
#pragma unroll
    for (int i = 0; i < 8; ++i) se[i] += __shfl_xor(se[i], m, 32);
#pragma unroll
  for (int i = 0; i < 8; ++i) {
    int row = rowbase + r0 + i + hi8;
    float inv = 1.0f / se[i];
    wOut[(size_t)row * N_ + laneR]      = e0[i] * inv;
    wOut[(size_t)row * N_ + 16 + laneR] = e1[i] * inv;
  }
}

// ------------------------------------------------------- per-variable GRNs
// One block = 128 rows (8 waves, one 16-row M-tile per wave). Weight panels
// async-staged once per block per n into LDS and shared by all waves.
// GLU, skip-add, LayerNorm (shfl_xor butterflies) and the softmax-weighted
// combine all run in registers.
__global__ void __launch_bounds__(256)
k_var(const _Float16* __restrict__ e16,
      const _Float16* __restrict__ vW1T, const _Float16* __restrict__ vWskipT,
      const _Float16* __restrict__ vW2T, const _Float16* __restrict__ vWgT,
      const float* __restrict__ vb1, const float* __restrict__ vbskip,
      const float* __restrict__ vb2, const float* __restrict__ vbg,
      const float* __restrict__ vg, const float* __restrict__ vbv,
      const float* __restrict__ wIn, float* __restrict__ outPtr) {
  __shared__ _Float16 sE[128 * D_];       // 16KB embedding slice [row][64]
  __shared__ _Float16 sAct[128 * H_];     // 32KB h1 then h2 (per-wave rows)
  __shared__ _Float16 sWa[128 * 128];     // 32KB: W1(16K)+Wskip(16K), reused for Wg gate half
  __shared__ _Float16 sW2[128 * H_];      // 32KB W2 panel [col][128]
  __shared__ _Float16 sWgA[128 * H_];     // 32KB Wg cols 0..127 [col][128]

  const int tid = threadIdx.x;
  const int wave = tid >> 5, lane = tid & 31;
  const int laneR = lane & 15, kh = (lane >> 4) & 1;
  const int hi8 = (lane & 16) ? 8 : 0;
  const int r0 = wave * 16;                    // wave's row tile within block
  const int rowbase = blockIdx.x * 128;
  const v8f Z = {0.f, 0.f, 0.f, 0.f, 0.f, 0.f, 0.f, 0.f};

  float outReg[64];                            // [ct][i] output accumulators
#pragma unroll
  for (int q = 0; q < 64; ++q) outReg[q] = 0.0f;
  float tv[64];                                // skip, then GLU result

  auto stageE = [&](int n) {
    for (int j = 0; j < 4; ++j) {              // 1024 chunks of 16B
      int c = tid + j * 256;
      int r = c >> 3, off = (c & 7) * 8;
      cp16(sE + c * 8, e16 + ((size_t)(rowbase + r) * N_ + n) * D_ + off);
    }
  };
  auto stageW1Ws = [&](int n) {                // two 16KB panels -> sWa
    const _Float16* g1 = vW1T + (size_t)n * H_ * D_;
    const _Float16* gs = vWskipT + (size_t)n * H_ * D_;
    for (int j = 0; j < 4; ++j) {
      int c = tid + j * 256;
      cp16(sWa + c * 8, g1 + c * 8);
      cp16(sWa + 8192 + c * 8, gs + c * 8);
    }
  };
  auto stageW2 = [&](int n) {                  // 32KB
    const _Float16* g = vW2T + (size_t)n * H_ * H_;
    for (int j = 0; j < 8; ++j) { int c = tid + j * 256; cp16(sW2 + c * 8, g + c * 8); }
  };
  auto stageWg = [&](int n) {                  // a-half -> sWgA, gate half -> sWa
    const _Float16* ga = vWgT + (size_t)n * 256 * H_;
    const _Float16* gg = ga + (size_t)128 * H_;
    for (int j = 0; j < 8; ++j) {
      int c = tid + j * 256;
      cp16(sWgA + c * 8, ga + c * 8);
      cp16(sWa + c * 8, gg + c * 8);
    }
  };

  stageE(0);
  stageW1Ws(0);

  for (int n = 0; n < N_; ++n) {
    wait_async();                              // E, W1, Wskip resident
    stageW2(n);                                // overlaps GEMM1

    // ---- GEMM1 (K=64): aW = E@W1, aS = E@Wskip, all 8 col-tiles ----
    v8f aW[8], aS[8];
#pragma unroll
    for (int ct = 0; ct < 8; ++ct) { aW[ct] = Z; aS[ct] = Z; }
#pragma unroll
    for (int k0 = 0; k0 < D_; k0 += 32) {
      v16h a = load_frag16(sE + (r0 + laneR) * D_ + k0 + kh * 8);
#pragma unroll
      for (int ct = 0; ct < 8; ++ct) {
        v16h b = load_frag16(sWa + (ct * 16 + laneR) * D_ + k0 + kh * 8);
        aW[ct] = WMMA16(a, b, aW[ct]);
        v16h b2 = load_frag16(sWa + 8192 + (ct * 16 + laneR) * D_ + k0 + kh * 8);
        aS[ct] = WMMA16(a, b2, aS[ct]);
      }
    }
#pragma unroll
    for (int ct = 0; ct < 8; ++ct) {
      int col = ct * 16 + laneR;
      float b1v = vb1[n * H_ + col], bsv = vbskip[n * H_ + col];
#pragma unroll
      for (int i = 0; i < 8; ++i) {
        int row = i + hi8;
        sAct[(r0 + row) * H_ + col] = (_Float16)elu_f(aW[ct][i] + b1v);
        tv[ct * 8 + i] = aS[ct][i] + bsv;      // skip, kept in registers
      }
    }
    wait_async();                              // W2 resident; waves done with W1/Ws
    stageWg(n);                                // overlaps GEMM2 (gate half -> sWa)

    // ---- GEMM2 (K=128): h2 = elu(h1) @ W2 + b2 (own rows, no barrier) ----
    v8f h2[8];
#pragma unroll
    for (int ct = 0; ct < 8; ++ct) h2[ct] = Z;
#pragma unroll
    for (int k0 = 0; k0 < H_; k0 += 32) {
      v16h a = load_frag16(sAct + (r0 + laneR) * H_ + k0 + kh * 8);
#pragma unroll
      for (int ct = 0; ct < 8; ++ct) {
        v16h b = load_frag16(sW2 + (ct * 16 + laneR) * H_ + k0 + kh * 8);
        h2[ct] = WMMA16(a, b, h2[ct]);
      }
    }
#pragma unroll
    for (int ct = 0; ct < 8; ++ct) {
      int col = ct * 16 + laneR;
      float b2v = vb2[n * H_ + col];
#pragma unroll
      for (int i = 0; i < 8; ++i)
        sAct[(r0 + i + hi8) * H_ + col] = (_Float16)(h2[ct][i] + b2v);
    }
    wait_async();                              // Wg halves resident

    // ---- GEMM3 pairs (K=128): a-part (sWgA) and gate (sWa), GLU in regs ----
#pragma unroll
    for (int ct = 0; ct < 8; ++ct) {
      v8f ga = Z, gg = Z;
#pragma unroll
      for (int k0 = 0; k0 < H_; k0 += 32) {
        v16h a = load_frag16(sAct + (r0 + laneR) * H_ + k0 + kh * 8);
        v16h bA = load_frag16(sWgA + (ct * 16 + laneR) * H_ + k0 + kh * 8);
        ga = WMMA16(a, bA, ga);
        v16h bG = load_frag16(sWa + (ct * 16 + laneR) * H_ + k0 + kh * 8);
        gg = WMMA16(a, bG, gg);
      }
      int col = ct * 16 + laneR;
      float bav = vbg[n * 256 + col], bgv = vbg[n * 256 + 128 + col];
#pragma unroll
      for (int i = 0; i < 8; ++i) {
        float av = ga[i] + bav;
        float gv = gg[i] + bgv;
        tv[ct * 8 + i] = av * sigmoid_f(gv) + tv[ct * 8 + i];   // GLU + skip
      }
    }

    // ---- LayerNorm over H=128: 16-lane shfl_xor butterflies ----
    float s1[8], s2[8];
#pragma unroll
    for (int i = 0; i < 8; ++i) { s1[i] = 0.f; s2[i] = 0.f; }
#pragma unroll
    for (int ct = 0; ct < 8; ++ct)
#pragma unroll
      for (int i = 0; i < 8; ++i) { float t = tv[ct * 8 + i]; s1[i] += t; s2[i] += t * t; }
#pragma unroll
    for (int m = 1; m < 16; m <<= 1)
#pragma unroll
      for (int i = 0; i < 8; ++i) {
        s1[i] += __shfl_xor(s1[i], m, 32);
        s2[i] += __shfl_xor(s2[i], m, 32);
      }

    // ---- normalize + softmax-weighted combine into registers ----
#pragma unroll
    for (int i = 0; i < 8; ++i) {
      float mu = s1[i] * (1.0f / H_);
      float var = s2[i] * (1.0f / H_) - mu * mu;
      float rs = rsqrtf(var + LN_EPS);
      int row = rowbase + r0 + i + hi8;
      float wsel = wIn[(size_t)row * N_ + n];
#pragma unroll
      for (int ct = 0; ct < 8; ++ct) {
        int col = ct * 16 + laneR;
        float val = (tv[ct * 8 + i] - mu) * rs * vg[n * H_ + col] + vbv[n * H_ + col];
        outReg[ct * 8 + i] += wsel * val;
      }
    }
    __syncthreads();                           // panels consumed; safe to restage
    if (n + 1 < N_) {                          // overlaps next-iteration latency
      stageE(n + 1);
      stageW1Ws(n + 1);
    }
  }

  // ---- epilogue: store out tile ----
#pragma unroll
  for (int ct = 0; ct < 8; ++ct)
#pragma unroll
    for (int i = 0; i < 8; ++i) {
      int row = rowbase + r0 + i + hi8;
      outPtr[(size_t)row * H_ + ct * 16 + laneR] = outReg[ct * 8 + i];
    }
}

// ----------------------------------------------------------------------------
extern "C" void kernel_launch(void* const* d_in, const int* in_sizes, int n_in,
                              void* d_out, int out_size, void* d_ws, size_t ws_size,
                              hipStream_t stream) {
  (void)in_sizes; (void)n_in; (void)out_size; (void)ws_size;

  const float* emb    = (const float*)d_in[0];
  const float* ctx    = (const float*)d_in[1];
  const float* fWskip = (const float*)d_in[2];
  const float* fbskip = (const float*)d_in[3];
  const float* fW1    = (const float*)d_in[4];
  const float* fb1    = (const float*)d_in[5];
  const float* fWc    = (const float*)d_in[6];
  const float* fW2    = (const float*)d_in[7];
  const float* fb2    = (const float*)d_in[8];
  const float* fWg    = (const float*)d_in[9];
  const float* fbg    = (const float*)d_in[10];
  const float* fg     = (const float*)d_in[11];
  const float* fbv    = (const float*)d_in[12];
  const float* vWskip = (const float*)d_in[13];
  const float* vbskip = (const float*)d_in[14];
  const float* vW1    = (const float*)d_in[15];
  const float* vb1    = (const float*)d_in[16];
  const float* vW2    = (const float*)d_in[17];
  const float* vb2    = (const float*)d_in[18];
  const float* vWg    = (const float*)d_in[19];
  const float* vbg    = (const float*)d_in[20];
  const float* vg     = (const float*)d_in[21];
  const float* vbv    = (const float*)d_in[22];

  // Workspace carve (f16 copies, transposed weights, ctx projection)
  char* w = (char*)d_ws;
  size_t off = 0;
  auto alloc = [&](size_t bytes) -> char* {
    char* p = w + off;
    off = (off + bytes + 255) & ~(size_t)255;
    return p;
  };
  _Float16* e16     = (_Float16*)alloc((size_t)BT_ * ND_ * 2);
  _Float16* fW1T    = (_Float16*)alloc((size_t)ND_ * H_ * 2);
  _Float16* fWskipT = (_Float16*)alloc((size_t)ND_ * N_ * 2);
  _Float16* fW2T    = (_Float16*)alloc((size_t)H_ * H_ * 2);
  _Float16* fWgT    = (_Float16*)alloc((size_t)H_ * 2 * N_ * 2);
  _Float16* vW1T    = (_Float16*)alloc((size_t)N_ * D_ * H_ * 2);
  _Float16* vWskipT = (_Float16*)alloc((size_t)N_ * D_ * H_ * 2);
  _Float16* vW2T    = (_Float16*)alloc((size_t)N_ * H_ * H_ * 2);
  _Float16* vWgT    = (_Float16*)alloc((size_t)N_ * H_ * 256 * 2);
  float*    ctxh    = (float*)alloc((size_t)B_ * H_ * 4);

  float* outPtr = (float*)d_out;                   // [BT, H]
  float* wOut   = outPtr + (size_t)BT_ * H_;       // [BT, N]

  // precision conversion + transposes
  k_cvt<<<4096, 256, 0, stream>>>(emb, e16, BT_ * ND_);
  k_cvtT<<<512, 256, 0, stream>>>(fW1, fW1T, 1, ND_, H_);
  k_cvtT<<<128, 256, 0, stream>>>(fWskip, fWskipT, 1, ND_, N_);
  k_cvtT<<<64, 256, 0, stream>>>(fW2, fW2T, 1, H_, H_);
  k_cvtT<<<32, 256, 0, stream>>>(fWg, fWgT, 1, H_, 2 * N_);
  k_cvtT<<<512, 256, 0, stream>>>(vW1, vW1T, N_, D_, H_);
  k_cvtT<<<512, 256, 0, stream>>>(vWskip, vWskipT, N_, D_, H_);
  k_cvtT<<<1024, 256, 0, stream>>>(vW2, vW2T, N_, H_, H_);
  k_cvtT<<<2048, 256, 0, stream>>>(vWg, vWgT, N_, H_, 256);
  k_ctx<<<B_, H_, 0, stream>>>(ctx, fWc, ctxh);

  // flatten GRN -> softmax weights (128 rows/block, 8 waves)
  k_flat<<<BT_ / 128, 256, 0, stream>>>(e16, fW1T, fWskipT, fW2T, fWgT,
                                        fb1, fbskip, fb2, fbg, fg, fbv,
                                        ctxh, wOut);

  // per-variable GRNs + weighted combine -> outputs (128 rows/block, 8 waves)
  k_var<<<BT_ / 128, 256, 0, stream>>>(e16, vW1T, vWskipT, vW2T, vWgT,
                                       vb1, vbskip, vb2, vbg, vg, vbv,
                                       wOut, outPtr);
}
